// MemoryAdapterLayer_47476568490236
// MI455X (gfx1250) — matrix operand order repaired and checked by reading, not compiled, not run on hardware
//
#include <hip/hip_runtime.h>
#include <stdint.h>

// ---------------------------------------------------------------------------
// MI455X (gfx1250) implementation of MemoryAdapterLayer.
// bf16 WMMA (v_wmma_f32_16x16x32_bf16) for all 4 GEMMs, fp32 softmax stats.
// 64x64 per-wave tiles (4x4 accumulators) for 1 b128-load/WMMA intensity.
// Batched grid.z when workspace allows; async-to-LDS staging in transpose
// via the gfx1250 global_load_async_to_lds_b128 builtin.
// ---------------------------------------------------------------------------

typedef __bf16 bf16_t;
typedef __attribute__((ext_vector_type(16))) __bf16 v16bf;
typedef __attribute__((ext_vector_type(8)))  __bf16 v8bf;
typedef __attribute__((ext_vector_type(8)))  float  v8f;
typedef int vi4 __attribute__((vector_size(16)));   // matches builtin param type

struct Frag2 { v8bf lo, hi; };

#if __has_builtin(__builtin_amdgcn_global_load_async_to_lds_b128)
#define HAVE_ASYNC_LDS 1
static __device__ inline void async_copy_b128(const void* gsrc, void* ldst) {
    __builtin_amdgcn_global_load_async_to_lds_b128((vi4*)gsrc, (vi4*)ldst, 0, 0);
}
static __device__ inline void wait_async0() {
#if __has_builtin(__builtin_amdgcn_s_wait_asynccnt)
    __builtin_amdgcn_s_wait_asynccnt(0);
#else
    asm volatile("s_wait_asynccnt 0" ::: "memory");
#endif
}
#endif

// A fragment (16x32 bf16): lane L -> row m = L%16; K-half = L/16.
// Per lane: K = half*8 .. half*8+7  and  16+half*8 .. 16+half*8+7
static __device__ inline v16bf load_a_frag(const bf16_t* __restrict__ base,
                                           int ld, int row, int k0, int half) {
    const bf16_t* p = base + (size_t)row * ld + k0 + half * 8;
    Frag2 f;
    f.lo = *(const v8bf*)p;
    f.hi = *(const v8bf*)(p + 16);
    return __builtin_bit_cast(v16bf, f);
}

// B fragment (32x16 bf16), fed from B^T row-major [N,K]:
// lane L -> col n = L%16; per lane K = half*8 .. half*8+7  and  16+half*8 .. 16+half*8+7
static __device__ inline v16bf load_b_frag(const bf16_t* __restrict__ base,
                                           int ld, int col, int k0, int half) {
    const bf16_t* p = base + (size_t)col * ld + k0 + half * 8;
    Frag2 f;
    f.lo = *(const v8bf*)p;
    f.hi = *(const v8bf*)(p + 16);
    return __builtin_bit_cast(v16bf, f);
}

static __device__ inline v8f vzero8() {
    v8f z;
#pragma unroll
    for (int i = 0; i < 8; ++i) z[i] = 0.0f;
    return z;
}

// Batched TN GEMM: out[z][M,N] = A[z][M,K] * (Btr[z][N,K])^T (+ bias[n]).
// 256 threads = 8 waves in a 2(M) x 4(N) grid; wave tile 64x64 (4x4 WMMA
// accumulators); block tile 128x256. All problem dims divide exactly.
template <bool ADD_BIAS, bool OUT_BF16>
__global__ __launch_bounds__(256)
void gemm_tn_wmma(const bf16_t* __restrict__ A, const bf16_t* __restrict__ Btr,
                  const float* __restrict__ bias, void* __restrict__ out,
                  int K, int lda, int ldb, int ldo,
                  size_t strideA, size_t strideB, size_t strideO) {
    const int lane = threadIdx.x & 31;
    const int wave = threadIdx.x >> 5;
    const int wm = wave & 1;        // 2 waves along M
    const int wn = wave >> 1;       // 4 waves along N
    const int m0 = blockIdx.x * 128 + wm * 64;
    const int n0 = blockIdx.y * 256 + wn * 64;
    const int half = lane >> 4;
    const int l16  = lane & 15;

    const bf16_t* Az = A + blockIdx.z * strideA;
    const bf16_t* Bz = Btr + blockIdx.z * strideB;

    v8f acc[4][4];
#pragma unroll
    for (int i = 0; i < 4; ++i)
#pragma unroll
        for (int j = 0; j < 4; ++j) acc[i][j] = vzero8();

    const bf16_t* arow = Az + (size_t)(m0 + l16) * lda;
    const bf16_t* brow = Bz + (size_t)(n0 + l16) * ldb;

    for (int k0 = 0; k0 < K; k0 += 32) {
        if (k0 + 32 < K) {   // gfx1250 global_prefetch of next K tile
            __builtin_prefetch(arow + k0 + 32 + half * 8, 0, 1);
            __builtin_prefetch(brow + k0 + 32 + half * 16, 0, 1);
        }
        v16bf af[4], bf[4];
#pragma unroll
        for (int mt = 0; mt < 4; ++mt)
            af[mt] = load_a_frag(Az, lda, m0 + mt * 16 + l16, k0, half);
#pragma unroll
        for (int nt = 0; nt < 4; ++nt)
            bf[nt] = load_b_frag(Bz, ldb, n0 + nt * 16 + l16, k0, half);
#pragma unroll
        for (int mt = 0; mt < 4; ++mt)
#pragma unroll
            for (int nt = 0; nt < 4; ++nt)
                acc[mt][nt] = __builtin_amdgcn_wmma_f32_16x16x32_bf16(
                    false, af[mt], false, bf[nt], (short)0, acc[mt][nt],
                    false, false);
    }

    // C/D layout: lane L -> col n = L%16; VGPR v -> row m = v + (L/16)*8
#pragma unroll
    for (int mt = 0; mt < 4; ++mt) {
#pragma unroll
        for (int nt = 0; nt < 4; ++nt) {
            const int col = n0 + nt * 16 + l16;
            const float bv = ADD_BIAS ? bias[col] : 0.0f;
#pragma unroll
            for (int v = 0; v < 8; ++v) {
                const int row = m0 + mt * 16 + v + half * 8;
                const float val = acc[mt][nt][v] + bv;
                if (OUT_BF16)
                    ((bf16_t*)out)[blockIdx.z * strideO + (size_t)row * ldo + col] =
                        (bf16_t)val;
                else
                    ((float*)out)[blockIdx.z * strideO + (size_t)row * ldo + col] =
                        val;
            }
        }
    }
}

// vectorized fp32 -> bf16 cast: 8 elements per thread (two b128 in, one out)
__global__ __launch_bounds__(256)
void cast_f32_bf16_v8(const float* __restrict__ in, bf16_t* __restrict__ out,
                      size_t n) {
    size_t base = ((size_t)blockIdx.x * blockDim.x + threadIdx.x) * 8;
    if (base >= n) return;
    float4 a = *(const float4*)(in + base);
    float4 b = *(const float4*)(in + base + 4);
    v8bf o;
    o[0] = (bf16_t)a.x; o[1] = (bf16_t)a.y; o[2] = (bf16_t)a.z; o[3] = (bf16_t)a.w;
    o[4] = (bf16_t)b.x; o[5] = (bf16_t)b.y; o[6] = (bf16_t)b.z; o[7] = (bf16_t)b.w;
    *(v8bf*)(out + base) = o;
}

// in: [B, rows, cols] fp32 -> out: [B, cols, rows] bf16.
// Stages the fp32 tile in LDS; uses gfx1250 async global->LDS when available.
__global__ __launch_bounds__(256)
void transpose_cast(const float* __restrict__ in, bf16_t* __restrict__ out,
                    int rows, int cols) {
    __shared__ float tile[32][36];   // pitch 36 floats: 16B-aligned rows
    const int b = blockIdx.z;
    const float* inb = in + (size_t)b * rows * cols;
    bf16_t* outb = out + (size_t)b * rows * cols;
    const int r0 = blockIdx.x * 32, c0 = blockIdx.y * 32;
    const int t = threadIdx.x;
    const int r = t >> 3, c4 = (t & 7) * 4;    // each thread: one 16B chunk
    const float* src = inb + (size_t)(r0 + r) * cols + (c0 + c4);
#ifdef HAVE_ASYNC_LDS
    async_copy_b128(src, &tile[r][c4]);
    wait_async0();
    __syncthreads();
#else
    *(float4*)&tile[r][c4] = *(const float4*)src;
    __syncthreads();
#endif
    const int tx = t & 31, ty = t >> 5;        // 8 cols per pass
#pragma unroll
    for (int i = 0; i < 32; i += 8)
        outb[(size_t)(c0 + ty + i) * rows + (r0 + tx)] = (bf16_t)tile[tx][ty + i];
}

// one block per row of M=4096; each thread owns M/256 = 16 elements which are
// kept in registers across the max / exp-sum / scale passes (single global
// read of the score row, single exp per element).
__global__ __launch_bounds__(256)
void softmax_rows(const float* __restrict__ scores, bf16_t* __restrict__ w, int M) {
    __shared__ float red[256];
    const int row = blockIdx.x;
    const int tid = threadIdx.x;
    const float* srow = scores + (size_t)row * M;
    bf16_t* wrow = w + (size_t)row * M;

    float v[16];                       // M/256 == 16 for this problem
    const int per = M >> 8;

    float mx = -3.402823466e38f;
#pragma unroll
    for (int i = 0; i < 16; ++i) {
        if (i < per) {
            v[i] = srow[tid + (i << 8)];
            mx = fmaxf(mx, v[i]);
        }
    }
    red[tid] = mx; __syncthreads();
    for (int s = 128; s > 0; s >>= 1) {
        if (tid < s) red[tid] = fmaxf(red[tid], red[tid + s]);
        __syncthreads();
    }
    mx = red[0]; __syncthreads();

    float sum = 0.0f;
#pragma unroll
    for (int i = 0; i < 16; ++i) {
        if (i < per) {
            v[i] = __expf(v[i] - mx);
            sum += v[i];
        }
    }
    red[tid] = sum; __syncthreads();
    for (int s = 128; s > 0; s >>= 1) {
        if (tid < s) red[tid] += red[tid + s];
        __syncthreads();
    }
    const float inv = 1.0f / red[0];
#pragma unroll
    for (int i = 0; i < 16; ++i)
        if (i < per) wrow[tid + (i << 8)] = (bf16_t)(v[i] * inv);
}

// ---------------------------------------------------------------------------

extern "C" void kernel_launch(void* const* d_in, const int* in_sizes, int n_in,
                              void* d_out, int out_size, void* d_ws, size_t ws_size,
                              hipStream_t stream) {
    const int B = 4, S = 2048, M = 4096, DQ = 1024, DM = 512;
    const int BS = B * S;

    const float* x      = (const float*)d_in[0];
    const float* memory = (const float*)d_in[1];
    const float* Wq     = (const float*)d_in[2];
    const float* bq     = (const float*)d_in[3];
    const float* Wm     = (const float*)d_in[4];
    const float* bm     = (const float*)d_in[5];

    // workspace layout (256B aligned)
    uint8_t* ws = (uint8_t*)d_ws;
    auto alloc = [&](size_t bytes) {
        uint8_t* p = ws;
        ws += (bytes + 255) & ~(size_t)255;
        return p;
    };
    bf16_t* xb   = (bf16_t*)alloc((size_t)BS * DQ * 2);
    bf16_t* memb = (bf16_t*)alloc((size_t)B * M * DM * 2);
    bf16_t* memT = (bf16_t*)alloc((size_t)B * M * DM * 2);
    bf16_t* Wqb  = (bf16_t*)alloc((size_t)DM * DQ * 2);
    bf16_t* Wmb  = (bf16_t*)alloc((size_t)DQ * DM * 2);
    bf16_t* qb   = (bf16_t*)alloc((size_t)BS * DM * 2);
    bf16_t* attb = (bf16_t*)alloc((size_t)BS * DM * 2);

    // scores/weights: all batches at once if workspace allows, else per-batch
    const size_t scoB = (size_t)S * M * 4, wgtB = (size_t)S * M * 2;
    const size_t used = (size_t)(ws - (uint8_t*)d_ws);
    const bool full = ws_size >= used + (size_t)B * (scoB + wgtB) + 4096;
    const int zspan = full ? B : 1;
    float*  sco = (float*) alloc(scoB * zspan);
    bf16_t* wgt = (bf16_t*)alloc(wgtB * zspan);

    // output = (x, transformed): copy x into first half
    (void)hipMemcpyAsync(d_out, x, (size_t)BS * DQ * sizeof(float),
                         hipMemcpyDeviceToDevice, stream);

    // casts (all sizes divisible by 8*256)
    {
        size_t n = (size_t)BS * DQ;
        cast_f32_bf16_v8<<<dim3(n / (8 * 256)), dim3(256), 0, stream>>>(x, xb, n);
        n = (size_t)B * M * DM;
        cast_f32_bf16_v8<<<dim3(n / (8 * 256)), dim3(256), 0, stream>>>(memory, memb, n);
        n = (size_t)DM * DQ;
        cast_f32_bf16_v8<<<dim3(n / (8 * 256)), dim3(256), 0, stream>>>(Wq, Wqb, n);
        n = (size_t)DQ * DM;
        cast_f32_bf16_v8<<<dim3(n / (8 * 256)), dim3(256), 0, stream>>>(Wm, Wmb, n);
        transpose_cast<<<dim3(M / 32, DM / 32, B), dim3(256), 0, stream>>>(
            memory, memT, M, DM);
    }

    // 1) query = x @ Wq^T + bq  -> qb bf16 [BS, DM]
    gemm_tn_wmma<true, true><<<dim3(BS / 128, DM / 256, 1), dim3(256), 0, stream>>>(
        xb, Wqb, bq, qb, DQ, DQ, DQ, DM, 0, 0, 0);

    // 2-4) scores -> softmax -> attended (batched over grid.z when possible)
    for (int b = 0; b < B; b += zspan) {
        const bf16_t* qbb = qb + (size_t)b * S * DM;
        const bf16_t* mbb = memb + (size_t)b * M * DM;
        const bf16_t* mtb = memT + (size_t)b * M * DM;
        bf16_t* atb = attb + (size_t)b * S * DM;

        // scores[s,m] = q[s,:] . memory[m,:]   (K = DM)
        gemm_tn_wmma<false, false>
            <<<dim3(S / 128, M / 256, zspan), dim3(256), 0, stream>>>(
                qbb, mbb, nullptr, sco, DM, DM, DM, M,
                (size_t)S * DM, (size_t)M * DM, (size_t)S * M);

        softmax_rows<<<dim3(S * zspan), dim3(256), 0, stream>>>(sco, wgt, M);

        // attended[s,d] = weights[s,:] . memory[:,d]  (K = M, B^T = memory^T)
        gemm_tn_wmma<false, true>
            <<<dim3(S / 128, DM / 256, zspan), dim3(256), 0, stream>>>(
                wgt, mtb, nullptr, atb, M, M, M, DM,
                (size_t)S * M, (size_t)M * DM, (size_t)S * DM);
    }

    // 5) transformed = attended @ Wm^T + bm -> second half of d_out (fp32)
    float* tout = (float*)d_out + (size_t)BS * DQ;
    gemm_tn_wmma<true, false><<<dim3(BS / 128, DQ / 256, 1), dim3(256), 0, stream>>>(
        attb, Wmb, bm, tout, DM, DM, DM, DQ, 0, 0, 0);
}
